// FixedLLaMAMoE_86904368268075
// MI455X (gfx1250) — matrix operand order repaired and checked by reading, not compiled, not run on hardware
//
#include <hip/hip_runtime.h>
#include <stdint.h>

// Problem constants (match reference)
#define NTOK 8192   // B*T
#define CDIM 2048
#define IDIM 1408
#define NEXP 8
#define KTOP 2

// GEMM tiling
#define BM 64
#define BN 128
#define BK 32
#define LDSP (BK + 8)   // 80B rows: 16B-aligned fragments, spread banks

typedef __bf16 v16bf __attribute__((ext_vector_type(16)));
typedef __bf16 bfx8  __attribute__((ext_vector_type(8)));
typedef __bf16 bfx4  __attribute__((ext_vector_type(4)));
typedef float  v8f   __attribute__((ext_vector_type(8)));

__device__ __forceinline__ v16bf frag_ld(const __bf16* p0, const __bf16* p1) {
  bfx8 lo = *(const bfx8*)p0;
  bfx8 hi = *(const bfx8*)p1;
  return __builtin_shufflevector(lo, hi, 0,1,2,3,4,5,6,7,8,9,10,11,12,13,14,15);
}

__device__ __forceinline__ v8f wmma_bf16(v16bf a, v16bf b, v8f c) {
  // D = A(16x32 bf16) * B(32x16 bf16) + C(16x16 f32)
  return __builtin_amdgcn_wmma_f32_16x16x32_bf16(false, a, false, b, (short)0, c,
                                                 false, false);
}

// CDNA5 async global->LDS DMA (ASYNCcnt-tracked), GV addressing mode.
// LDS dest offset = low 32 bits of the generic pointer (LDS aperture maps
// addr[31:0] -> LDS offset, ISA 10.2).
__device__ __forceinline__ void async_copy_b128(void* lds_ptr, const void* gptr) {
  uint32_t lds_off = (uint32_t)(uintptr_t)lds_ptr;
  uint64_t gaddr   = (uint64_t)(uintptr_t)gptr;
  asm volatile("global_load_async_to_lds_b128 %0, %1, off"
               :
               : "v"(lds_off), "v"(gaddr)
               : "memory");
}
__device__ __forceinline__ void wait_async0() {
  asm volatile("s_wait_asynccnt 0" ::: "memory");
}

__device__ __forceinline__ float fast_silu_mul(float a1, float a3) {
  // silu(a1) * a3 with v_rcp_f32 instead of a precise divide chain
  float r = __builtin_amdgcn_rcpf(1.f + __expf(-a1));
  return a1 * r * a3;
}

// ---------------------------------------------------------------------------
// 1) Router: logits = x @ gate_w, top-2, softmax over the 2 (f32), count/expert
// ---------------------------------------------------------------------------
__global__ __launch_bounds__(256) void moe_router(
    const float* __restrict__ x, const float* __restrict__ gate_w,
    int* __restrict__ topI, float* __restrict__ topP, int* __restrict__ cnt)
{
  int wave = threadIdx.x >> 5, lane = threadIdx.x & 31;
  int tok = blockIdx.x * 8 + wave;
  if (tok >= NTOK) return;
  const float* xr = x + (size_t)tok * CDIM;
  float acc[NEXP];
  #pragma unroll
  for (int e = 0; e < NEXP; ++e) acc[e] = 0.f;
  for (int c = lane; c < CDIM; c += 32) {
    float xv = xr[c];
    const float* g = gate_w + (size_t)c * NEXP;
    #pragma unroll
    for (int e = 0; e < NEXP; ++e) acc[e] += xv * g[e];
  }
  #pragma unroll
  for (int e = 0; e < NEXP; ++e) {
    #pragma unroll
    for (int off = 16; off > 0; off >>= 1)
      acc[e] += __shfl_xor(acc[e], off, 32);
  }
  if (lane == 0) {
    int i0 = 0; float l0 = acc[0];
    #pragma unroll
    for (int e = 1; e < NEXP; ++e) if (acc[e] > l0) { l0 = acc[e]; i0 = e; }
    int i1 = (i0 == 0) ? 1 : 0; float l1 = acc[i1];
    #pragma unroll
    for (int e = 0; e < NEXP; ++e)
      if (e != i0 && acc[e] > l1) { l1 = acc[e]; i1 = e; }
    float ex  = __expf(l1 - l0);
    float inv = 1.f / (1.f + ex);
    topI[2 * tok]     = i0;
    topI[2 * tok + 1] = i1;
    topP[2 * tok]     = inv;
    topP[2 * tok + 1] = ex * inv;
    atomicAdd(&cnt[i0], 1);
    atomicAdd(&cnt[i1], 1);
  }
}

// 2) Prefix offsets over 8 experts (tiny, 1 thread)
__global__ void moe_offsets(const int* __restrict__ cnt, int* __restrict__ offs,
                            int* __restrict__ cursor)
{
  int s = 0;
  for (int e = 0; e < NEXP; ++e) { offs[e] = s; cursor[e] = s; s += cnt[e]; }
}

// 3) Placement: compact (token,slot) pairs per expert. pid = token*2 + slot.
__global__ __launch_bounds__(256) void moe_place(
    const int* __restrict__ topI, int* __restrict__ cursor,
    int* __restrict__ tokList)
{
  int n = blockIdx.x * 256 + threadIdx.x;
  if (n >= NTOK) return;
  #pragma unroll
  for (int k = 0; k < KTOP; ++k) {
    int e = topI[2 * n + k];
    int pos = atomicAdd(&cursor[e], 1);
    tokList[pos] = 2 * n + k;
  }
}

// 4a) Elementwise f32 -> bf16
__global__ __launch_bounds__(256) void cvt_bf16(
    const float* __restrict__ s, __bf16* __restrict__ d, size_t n)
{
  size_t i = ((size_t)blockIdx.x * 256 + threadIdx.x) * 4;
  if (i + 3 >= n) return;
  float4 v = *(const float4*)(s + i);
  bfx4 o;
  o[0] = (__bf16)v.x; o[1] = (__bf16)v.y; o[2] = (__bf16)v.z; o[3] = (__bf16)v.w;
  *(bfx4*)(d + i) = o;
}

// 4b) Per-expert transpose + convert: src [E][R][Cc] f32 -> dst [E][Cc][R] bf16
__global__ __launch_bounds__(256) void transpose_cvt(
    const float* __restrict__ src, __bf16* __restrict__ dst, int R, int Cc)
{
  __shared__ float t[32][33];
  int e  = blockIdx.z;
  int c0 = blockIdx.x * 32, r0 = blockIdx.y * 32;
  const float* s = src + (size_t)e * R * Cc;
  __bf16*      d = dst + (size_t)e * R * Cc;
  int tx = threadIdx.x, ty = threadIdx.y;
  #pragma unroll
  for (int j = 0; j < 32; j += 8)
    t[ty + j][tx] = s[(size_t)(r0 + ty + j) * Cc + (c0 + tx)];
  __syncthreads();
  #pragma unroll
  for (int j = 0; j < 32; j += 8)
    d[(size_t)(c0 + ty + j) * R + (r0 + tx)] = (__bf16)t[tx][ty + j];
}

// ---------------------------------------------------------------------------
// 5) GEMM1 + SwiGLU: H[rows,I] = silu(Xg @ w1) * (Xg @ w3), Xg gathered tokens
//    A gathered from xb; w1t/w3t stored K-major: [E][I][C].
//    Double-buffered LDS, tiles staged with async global->LDS DMA.
// ---------------------------------------------------------------------------
__global__ __launch_bounds__(256) void moe_gemm1_swiglu(
    const __bf16* __restrict__ xb,
    const __bf16* __restrict__ w1t, const __bf16* __restrict__ w3t,
    const int* __restrict__ tokList, const int* __restrict__ cnt,
    const int* __restrict__ offs, __bf16* __restrict__ H)
{
  int e = blockIdx.z;
  int mcount = cnt[e];
  int m0 = blockIdx.y * BM;
  if (m0 >= mcount) return;
  int n0 = blockIdx.x * BN;
  int base = offs[e];

  __shared__ __bf16 As[2][BM][LDSP];
  __shared__ __bf16 B1s[2][BN][LDSP];
  __shared__ __bf16 B3s[2][BN][LDSP];
  __shared__ int rowTok[BM];

  int tid = threadIdx.x;
  if (tid < BM) {
    int gr = m0 + tid;
    if (gr >= mcount) gr = mcount - 1;   // rows past count masked on store
    rowTok[tid] = tokList[base + gr] >> 1;
  }
  __syncthreads();

  int lane = tid & 31, wave = tid >> 5;
  int wm = wave & 1, wn = wave >> 1;          // wave tile: 32(M) x 32(N)
  int laneRow = lane & 15, laneHi = lane >> 4;

  int aRow = tid >> 2, aCol = (tid & 3) * 8;  // 64 rows * 32 cols / 256 thr
  int bRow = tid >> 1, bCol = (tid & 1) * 16; // 128 rows * 32 cols / 256 thr
  const __bf16* aSrc = xb + (size_t)rowTok[aRow] * CDIM + aCol;
  size_t bOff = ((size_t)e * IDIM + (n0 + bRow)) * CDIM + bCol;

  v8f acc1[2][2] = {};
  v8f acc3[2][2] = {};

  auto issue_tile = [&](int buf, int k0) {
    async_copy_b128(&As[buf][aRow][aCol], aSrc + k0);
    const __bf16* b1 = w1t + bOff + k0;
    const __bf16* b3 = w3t + bOff + k0;
    async_copy_b128(&B1s[buf][bRow][bCol],     b1);
    async_copy_b128(&B1s[buf][bRow][bCol + 8], b1 + 8);
    async_copy_b128(&B3s[buf][bRow][bCol],     b3);
    async_copy_b128(&B3s[buf][bRow][bCol + 8], b3 + 8);
  };

  auto compute_tile = [&](int buf) {
    v16bf af[2];
    #pragma unroll
    for (int mt = 0; mt < 2; ++mt) {
      int r  = wm * 32 + mt * 16 + laneRow;
      int kh = laneHi * 8;
      af[mt] = frag_ld(&As[buf][r][kh], &As[buf][r][16 + kh]);
    }
    #pragma unroll
    for (int nt = 0; nt < 2; ++nt) {
      int nn = wn * 32 + nt * 16 + laneRow;
      int kb = laneHi * 8;
      v16bf b1f = frag_ld(&B1s[buf][nn][kb], &B1s[buf][nn][16 + kb]);
      v16bf b3f = frag_ld(&B3s[buf][nn][kb], &B3s[buf][nn][16 + kb]);
      #pragma unroll
      for (int mt = 0; mt < 2; ++mt) {
        acc1[mt][nt] = wmma_bf16(af[mt], b1f, acc1[mt][nt]);
        acc3[mt][nt] = wmma_bf16(af[mt], b3f, acc3[mt][nt]);
      }
    }
  };

  const int NIT = CDIM / BK;  // 64
  issue_tile(0, 0);
  wait_async0();
  __syncthreads();
  for (int it = 0; it < NIT; ++it) {
    if (it + 1 < NIT) issue_tile((it + 1) & 1, (it + 1) * BK);  // overlap DMA
    compute_tile(it & 1);
    wait_async0();      // next tile landed (own wave's DMA)
    __syncthreads();    // all waves' DMA landed; safe to read / overwrite
  }

  int wrow0 = m0 + wm * 32, wcol0 = n0 + wn * 32;
  #pragma unroll
  for (int mt = 0; mt < 2; ++mt) {
    #pragma unroll
    for (int nt = 0; nt < 2; ++nt) {
      #pragma unroll
      for (int i = 0; i < 8; ++i) {
        int r = wrow0 + mt * 16 + laneHi * 8 + i;
        if (r < mcount) {
          int cc = wcol0 + nt * 16 + laneRow;
          float h = fast_silu_mul(acc1[mt][nt][i], acc3[mt][nt][i]);
          H[(size_t)(base + r) * IDIM + cc] = (__bf16)h;
        }
      }
    }
  }
}

// ---------------------------------------------------------------------------
// 6) GEMM2: yPair[pid, C] = H[rows,I] @ w2[e]   (w2t stored [E][C][I], K-major)
//    No atomics: each (token,slot) writes its own row -> deterministic combine.
// ---------------------------------------------------------------------------
__global__ __launch_bounds__(256) void moe_gemm2(
    const __bf16* __restrict__ H, const __bf16* __restrict__ w2t,
    const int* __restrict__ tokList, const int* __restrict__ cnt,
    const int* __restrict__ offs, float* __restrict__ yPair)
{
  int e = blockIdx.z;
  int mcount = cnt[e];
  int m0 = blockIdx.y * BM;
  if (m0 >= mcount) return;
  int n0 = blockIdx.x * BN;
  int base = offs[e];

  __shared__ __bf16 As[2][BM][LDSP];
  __shared__ __bf16 Bs[2][BN][LDSP];
  __shared__ int pidS[BM];

  int tid = threadIdx.x;
  if (tid < BM) {
    int gr = m0 + tid;
    if (gr >= mcount) gr = mcount - 1;
    pidS[tid] = tokList[base + gr];
  }

  int lane = tid & 31, wave = tid >> 5;
  int wm = wave & 1, wn = wave >> 1;
  int laneRow = lane & 15, laneHi = lane >> 4;

  int aRow = tid >> 2, aCol = (tid & 3) * 8;
  int gra = m0 + aRow; if (gra >= mcount) gra = mcount - 1;
  const __bf16* aSrc = H + (size_t)(base + gra) * IDIM + aCol;
  int bRow = tid >> 1, bCol = (tid & 1) * 16;
  const __bf16* bSrc = w2t + ((size_t)e * CDIM + (n0 + bRow)) * IDIM + bCol;

  v8f acc[2][2] = {};

  auto issue_tile = [&](int buf, int k0) {
    async_copy_b128(&As[buf][aRow][aCol],     aSrc + k0);
    async_copy_b128(&Bs[buf][bRow][bCol],     bSrc + k0);
    async_copy_b128(&Bs[buf][bRow][bCol + 8], bSrc + k0 + 8);
  };

  auto compute_tile = [&](int buf) {
    v16bf af[2];
    #pragma unroll
    for (int mt = 0; mt < 2; ++mt) {
      int r  = wm * 32 + mt * 16 + laneRow;
      int kh = laneHi * 8;
      af[mt] = frag_ld(&As[buf][r][kh], &As[buf][r][16 + kh]);
    }
    #pragma unroll
    for (int nt = 0; nt < 2; ++nt) {
      int nn = wn * 32 + nt * 16 + laneRow;
      int kb = laneHi * 8;
      v16bf bf = frag_ld(&Bs[buf][nn][kb], &Bs[buf][nn][16 + kb]);
      #pragma unroll
      for (int mt = 0; mt < 2; ++mt)
        acc[mt][nt] = wmma_bf16(af[mt], bf, acc[mt][nt]);
    }
  };

  const int NIT = IDIM / BK;  // 44
  issue_tile(0, 0);
  wait_async0();
  __syncthreads();
  for (int it = 0; it < NIT; ++it) {
    if (it + 1 < NIT) issue_tile((it + 1) & 1, (it + 1) * BK);
    compute_tile(it & 1);
    wait_async0();
    __syncthreads();
  }

  int wrow0 = wm * 32, wcol0 = n0 + wn * 32;
  #pragma unroll
  for (int mt = 0; mt < 2; ++mt) {
    #pragma unroll
    for (int nt = 0; nt < 2; ++nt) {
      #pragma unroll
      for (int i = 0; i < 8; ++i) {
        int rl = wrow0 + mt * 16 + laneHi * 8 + i;    // row within block
        if (m0 + rl < mcount) {
          int pid = pidS[rl];
          int cc  = wcol0 + nt * 16 + laneRow;
          yPair[(size_t)pid * CDIM + cc] = acc[mt][nt][i];
        }
      }
    }
  }
}

// 7) Combine: y[n,c] = p0 * yPair[2n,c] + p1 * yPair[2n+1,c]  (fixed order)
__global__ __launch_bounds__(256) void moe_combine(
    const float* __restrict__ yPair, const float* __restrict__ topP,
    float* __restrict__ y)
{
  size_t i = ((size_t)blockIdx.x * 256 + threadIdx.x) * 4;
  size_t n = i / CDIM;
  size_t c = i % CDIM;
  float p0 = topP[2 * n], p1 = topP[2 * n + 1];
  float4 a = *(const float4*)(yPair + (2 * n) * (size_t)CDIM + c);
  float4 b = *(const float4*)(yPair + (2 * n + 1) * (size_t)CDIM + c);
  float4 o;
  o.x = p0 * a.x + p1 * b.x;
  o.y = p0 * a.y + p1 * b.y;
  o.z = p0 * a.z + p1 * b.z;
  o.w = p0 * a.w + p1 * b.w;
  *(float4*)(y + i) = o;
}

// ---------------------------------------------------------------------------
extern "C" void kernel_launch(void* const* d_in, const int* in_sizes, int n_in,
                              void* d_out, int out_size, void* d_ws, size_t ws_size,
                              hipStream_t stream)
{
  (void)in_sizes; (void)n_in; (void)out_size; (void)ws_size;
  const float* x  = (const float*)d_in[0];   // [B,T,C]
  const float* gw = (const float*)d_in[1];   // [C,E]
  const float* w1 = (const float*)d_in[2];   // [E,C,I]
  const float* w3 = (const float*)d_in[3];   // [E,C,I]
  const float* w2 = (const float*)d_in[4];   // [E,I,C]
  float* y = (float*)d_out;                  // [B,T,C]

  char* p = (char*)d_ws;
  auto alloc = [&](size_t bytes) {
    char* r = p;
    p += (bytes + 255) & ~(size_t)255;
    return r;
  };
  __bf16* xb   = (__bf16*)alloc((size_t)NTOK * CDIM * 2);
  __bf16* w1t  = (__bf16*)alloc((size_t)NEXP * IDIM * CDIM * 2);   // [E][I][C]
  __bf16* w3t  = (__bf16*)alloc((size_t)NEXP * IDIM * CDIM * 2);   // [E][I][C]
  __bf16* w2t  = (__bf16*)alloc((size_t)NEXP * CDIM * IDIM * 2);   // [E][C][I]
  __bf16* Hbuf = (__bf16*)alloc((size_t)NTOK * KTOP * IDIM * 2);
  float* yPair = (float*)alloc((size_t)NTOK * KTOP * CDIM * 4);
  int*   topI  = (int*)alloc((size_t)NTOK * KTOP * 4);
  float* topP  = (float*)alloc((size_t)NTOK * KTOP * 4);
  int* tokList = (int*)alloc((size_t)NTOK * KTOP * 4);
  int* cnt     = (int*)alloc(NEXP * 4);
  int* offs    = (int*)alloc(NEXP * 4);
  int* cursor  = (int*)alloc(NEXP * 4);

  hipMemsetAsync(cnt, 0, NEXP * sizeof(int), stream);

  moe_router<<<NTOK / 8, 256, 0, stream>>>(x, gw, topI, topP, cnt);
  moe_offsets<<<1, 1, 0, stream>>>(cnt, offs, cursor);
  moe_place<<<NTOK / 256, 256, 0, stream>>>(topI, cursor, tokList);

  cvt_bf16<<<(unsigned)((size_t)NTOK * CDIM / 4 / 256), 256, 0, stream>>>(
      x, xb, (size_t)NTOK * CDIM);
  dim3 tb(32, 8);
  transpose_cvt<<<dim3(IDIM / 32, CDIM / 32, NEXP), tb, 0, stream>>>(w1, w1t, CDIM, IDIM);
  transpose_cvt<<<dim3(IDIM / 32, CDIM / 32, NEXP), tb, 0, stream>>>(w3, w3t, CDIM, IDIM);
  transpose_cvt<<<dim3(CDIM / 32, IDIM / 32, NEXP), tb, 0, stream>>>(w2, w2t, IDIM, CDIM);

  moe_gemm1_swiglu<<<dim3(IDIM / BN, NTOK / BM, NEXP), 256, 0, stream>>>(
      xb, w1t, w3t, tokList, cnt, offs, Hbuf);
  moe_gemm2<<<dim3(CDIM / BN, NTOK / BM, NEXP), 256, 0, stream>>>(
      Hbuf, w2t, tokList, cnt, offs, yPair);
  moe_combine<<<(unsigned)((size_t)NTOK * CDIM / 4 / 256), 256, 0, stream>>>(
      yPair, topP, y);
}
